// SpGraphAttentionLayer_7627861917709
// MI455X (gfx1250) — compile-verified, matrix operand order checked by reading, not run on hardware
//
#include <hip/hip_runtime.h>
#include <hip/hip_bf16.h>
#include <stdint.h>

#define NN   50000
#define EE   800000
#define INC  256
#define OUTC 128
#define H1   64
#define H2   32

typedef __attribute__((ext_vector_type(16))) __bf16 v16bf;
typedef __attribute__((ext_vector_type(8)))  __bf16 v8bf;
typedef __attribute__((ext_vector_type(8)))  float  v8f;
typedef __attribute__((ext_vector_type(4)))  float  v4f;

__device__ __forceinline__ float lrelu(float v) { return v > 0.f ? v : 0.2f * v; }
__device__ __forceinline__ int   clampi(int v, int hi) { v = v < 0 ? 0 : v; return v >= hi ? hi - 1 : v; }
__device__ __forceinline__ v16bf cat8(v8bf lo, v8bf hi) {
  return __builtin_shufflevector(lo, hi, 0,1,2,3,4,5,6,7,8,9,10,11,12,13,14,15);
}

// ---------------------------------------------------------------- zero scratch
__global__ void zero_kernel(float* __restrict__ p, unsigned n) {
  unsigned i = blockIdx.x * blockDim.x + threadIdx.x;
  if (i < n) p[i] = 0.f;
}

// ------------------------------------------- bf16 conversion + weight transpose
// xb[N][256] = bf16(x); wT[128][256] = w^T; a1T[64][128] = a1_w^T; a2T[32][64] = a2_w^T
__global__ void prep_kernel(const float* __restrict__ x, const float* __restrict__ w,
                            const float* __restrict__ a1w, const float* __restrict__ a2w,
                            __bf16* __restrict__ xb, __bf16* __restrict__ wT,
                            __bf16* __restrict__ a1T, __bf16* __restrict__ a2T) {
  size_t i = (size_t)blockIdx.x * blockDim.x + threadIdx.x;
  const size_t XT = (size_t)NN * INC;
  if (i < XT) { xb[i] = (__bf16)x[i]; return; }
  size_t j = i - XT;
  if (j < (size_t)INC * OUTC) {           // w: [256][128] row-major -> wT[n][k]
    int k = (int)(j >> 7), n = (int)(j & 127);
    wT[(size_t)n * INC + k] = (__bf16)w[j];
    return;
  }
  j -= (size_t)INC * OUTC;
  if (j < (size_t)OUTC * H1) {            // a1_w: [128][64] -> a1T[n][k]
    int k = (int)(j >> 6), n = (int)(j & 63);
    a1T[(size_t)n * OUTC + k] = (__bf16)a1w[j];
    return;
  }
  j -= (size_t)OUTC * H1;
  if (j < (size_t)H1 * H2) {              // a2_w: [64][32] -> a2T[n][k]
    int k = (int)(j >> 5), n = (int)(j & 31);
    a2T[(size_t)n * H1 + k] = (__bf16)a2w[j];
  }
}

// ---------------------------------------------------------------- h = x @ w
// one wave per 16x16 output tile, K=256 in 8 WMMA steps
__global__ void node_gemm_kernel(const __bf16* __restrict__ xb, const __bf16* __restrict__ wT,
                                 float* __restrict__ h) {
  const int lane = threadIdx.x & 31;
  const int wave = blockIdx.x * (blockDim.x >> 5) + (threadIdx.x >> 5);
  const int mt = wave >> 3;                 // 3125 row tiles
  const int nt = wave & 7;                  // 8 col tiles
  const int row = mt * 16 + (lane & 15);
  const int col = nt * 16 + (lane & 15);
  const int sub = (lane >> 4) << 3;
  const __bf16* ap = xb + (size_t)row * INC;
  const __bf16* bp = wT + (size_t)col * INC;
  v8f acc = {};
  #pragma unroll
  for (int kk = 0; kk < INC; kk += 32) {
    v16bf a = cat8(*(const v8bf*)(ap + kk + sub), *(const v8bf*)(ap + kk + 16 + sub));
    v16bf b = cat8(*(const v8bf*)(bp + kk + sub), *(const v8bf*)(bp + kk + 16 + sub));
    acc = __builtin_amdgcn_wmma_f32_16x16x32_bf16(false, a, false, b, (short)0, acc, false, false);
  }
  #pragma unroll
  for (int r = 0; r < 8; r++) {
    const int m = mt * 16 + sub + r;
    h[(size_t)m * OUTC + nt * 16 + (lane & 15)] = acc[r];
  }
}

// ------------------------- layer1: z1 = |h[src]-h[dst]| @ a1_w + b1 ; BN stats
// one wave per 16-edge tile; 4 k-steps x 4 n-tiles of WMMA
__global__ void edge_l1_kernel(const float* __restrict__ h,
                               const int* __restrict__ srcv, const int* __restrict__ dstv,
                               const __bf16* __restrict__ a1T, const float* __restrict__ a1b,
                               float* __restrict__ z1,
                               float* __restrict__ gsum, float* __restrict__ gsq) {
  __shared__ float ssum[H1], ssq[H1];
  for (int i = threadIdx.x; i < H1; i += blockDim.x) { ssum[i] = 0.f; ssq[i] = 0.f; }
  __syncthreads();

  const int lane = threadIdx.x & 31;
  const int tile = blockIdx.x * (blockDim.x >> 5) + (threadIdx.x >> 5);
  const int e    = tile * 16 + (lane & 15);
  const int s = clampi(srcv[e], NN);
  const int d = clampi(dstv[e], NN);
  const int sub = (lane >> 4) << 3;
  const float* hs = h + (size_t)s * OUTC;
  const float* hd = h + (size_t)d * OUTC;

  v8f acc[4] = {};
  #pragma unroll
  for (int kk = 0; kk < OUTC; kk += 32) {
    v4f s0 = *(const v4f*)(hs + kk + sub);
    v4f s1 = *(const v4f*)(hs + kk + sub + 4);
    v4f s2 = *(const v4f*)(hs + kk + sub + 16);
    v4f s3 = *(const v4f*)(hs + kk + sub + 20);
    v4f t0 = *(const v4f*)(hd + kk + sub);
    v4f t1 = *(const v4f*)(hd + kk + sub + 4);
    v4f t2 = *(const v4f*)(hd + kk + sub + 16);
    v4f t3 = *(const v4f*)(hd + kk + sub + 20);
    v16bf a;
    #pragma unroll
    for (int j = 0; j < 4; j++) {
      a[j]      = (__bf16)fabsf(s0[j] - t0[j]);
      a[4 + j]  = (__bf16)fabsf(s1[j] - t1[j]);
      a[8 + j]  = (__bf16)fabsf(s2[j] - t2[j]);
      a[12 + j] = (__bf16)fabsf(s3[j] - t3[j]);
    }
    #pragma unroll
    for (int nt = 0; nt < 4; nt++) {
      const __bf16* bp = a1T + (size_t)(nt * 16 + (lane & 15)) * OUTC + kk + sub;
      v16bf b = cat8(*(const v8bf*)bp, *(const v8bf*)(bp + 16));
      acc[nt] = __builtin_amdgcn_wmma_f32_16x16x32_bf16(false, a, false, b, (short)0, acc[nt], false, false);
    }
  }
  #pragma unroll
  for (int nt = 0; nt < 4; nt++) {
    const int col = nt * 16 + (lane & 15);
    const float bias = a1b[col];
    float ps = 0.f, pq = 0.f;
    #pragma unroll
    for (int r = 0; r < 8; r++) {
      const int erow = tile * 16 + sub + r;
      const float v = acc[nt][r] + bias;
      z1[(size_t)erow * H1 + col] = v;
      ps += v; pq += v * v;
    }
    atomicAdd(&ssum[col], ps);
    atomicAdd(&ssq[col], pq);
  }
  __syncthreads();
  for (int i = threadIdx.x; i < H1; i += blockDim.x) {
    atomicAdd(&gsum[i], ssum[i]);
    atomicAdd(&gsq[i], ssq[i]);
  }
}

// ------------------------------------------------ BN stats -> scale/shift
__global__ void bn_finalize_kernel(const float* __restrict__ sum, const float* __restrict__ sq,
                                   const float* __restrict__ g, const float* __restrict__ b,
                                   float* __restrict__ scale, float* __restrict__ shift,
                                   int C, float invE) {
  int c = threadIdx.x;
  if (c >= C) return;
  float m   = sum[c] * invE;
  float var = sq[c] * invE - m * m;
  float sc  = g[c] * rsqrtf(var + 1e-5f);
  scale[c] = sc;
  shift[c] = b[c] - m * sc;
}

// ---------- layer2: z2 = lrelu(bn(z1)) @ a2_w + b2 ; BN stats (K=64, N=32)
__global__ void edge_l2_kernel(const float* __restrict__ z1,
                               const float* __restrict__ scale1, const float* __restrict__ shift1,
                               const __bf16* __restrict__ a2T, const float* __restrict__ a2b,
                               float* __restrict__ z2,
                               float* __restrict__ gsum, float* __restrict__ gsq) {
  __shared__ float ssum[H2], ssq[H2];
  for (int i = threadIdx.x; i < H2; i += blockDim.x) { ssum[i] = 0.f; ssq[i] = 0.f; }
  __syncthreads();

  const int lane = threadIdx.x & 31;
  const int tile = blockIdx.x * (blockDim.x >> 5) + (threadIdx.x >> 5);
  const int e    = tile * 16 + (lane & 15);
  const int sub  = (lane >> 4) << 3;
  const float* zr = z1 + (size_t)e * H1;

  v8f acc[2] = {};
  #pragma unroll
  for (int kk = 0; kk < H1; kk += 32) {
    v4f z0 = *(const v4f*)(zr + kk + sub);
    v4f z1v = *(const v4f*)(zr + kk + sub + 4);
    v4f z2v = *(const v4f*)(zr + kk + sub + 16);
    v4f z3v = *(const v4f*)(zr + kk + sub + 20);
    v4f c0 = *(const v4f*)(scale1 + kk + sub);
    v4f c1 = *(const v4f*)(scale1 + kk + sub + 4);
    v4f c2 = *(const v4f*)(scale1 + kk + sub + 16);
    v4f c3 = *(const v4f*)(scale1 + kk + sub + 20);
    v4f f0 = *(const v4f*)(shift1 + kk + sub);
    v4f f1 = *(const v4f*)(shift1 + kk + sub + 4);
    v4f f2 = *(const v4f*)(shift1 + kk + sub + 16);
    v4f f3 = *(const v4f*)(shift1 + kk + sub + 20);
    v16bf a;
    #pragma unroll
    for (int j = 0; j < 4; j++) {
      a[j]      = (__bf16)lrelu(z0[j]  * c0[j] + f0[j]);
      a[4 + j]  = (__bf16)lrelu(z1v[j] * c1[j] + f1[j]);
      a[8 + j]  = (__bf16)lrelu(z2v[j] * c2[j] + f2[j]);
      a[12 + j] = (__bf16)lrelu(z3v[j] * c3[j] + f3[j]);
    }
    #pragma unroll
    for (int nt = 0; nt < 2; nt++) {
      const __bf16* bp = a2T + (size_t)(nt * 16 + (lane & 15)) * H1 + kk + sub;
      v16bf b = cat8(*(const v8bf*)bp, *(const v8bf*)(bp + 16));
      acc[nt] = __builtin_amdgcn_wmma_f32_16x16x32_bf16(false, a, false, b, (short)0, acc[nt], false, false);
    }
  }
  #pragma unroll
  for (int nt = 0; nt < 2; nt++) {
    const int col = nt * 16 + (lane & 15);
    const float bias = a2b[col];
    float ps = 0.f, pq = 0.f;
    #pragma unroll
    for (int r = 0; r < 8; r++) {
      const int erow = tile * 16 + sub + r;
      const float v = acc[nt][r] + bias;
      z2[(size_t)erow * H2 + col] = v;
      ps += v; pq += v * v;
    }
    atomicAdd(&ssum[col], ps);
    atomicAdd(&ssq[col], pq);
  }
  __syncthreads();
  for (int i = threadIdx.x; i < H2; i += blockDim.x) {
    atomicAdd(&gsum[i], ssum[i]);
    atomicAdd(&gsq[i], ssq[i]);
  }
}

// -------- layer3 + score: edge_e = exp(-lrelu(lrelu(bn(z2)) @ a3 + b3)) + I(src==dst)
__global__ void edge_score_kernel(const float* __restrict__ z2,
                                  const int* __restrict__ srcv, const int* __restrict__ dstv,
                                  const float* __restrict__ scale2, const float* __restrict__ shift2,
                                  const float* __restrict__ a3w, const float* __restrict__ a3b,
                                  float* __restrict__ eweight, float* __restrict__ rowsum) {
  int e = blockIdx.x * blockDim.x + threadIdx.x;
  if (e >= EE) return;
  float accv = 0.f;
  const float* zr = z2 + (size_t)e * H2;
  #pragma unroll
  for (int c = 0; c < H2; c += 4) {
    v4f z  = *(const v4f*)(zr + c);
    v4f sc = *(const v4f*)(scale2 + c);
    v4f sh = *(const v4f*)(shift2 + c);
    v4f aw = *(const v4f*)(a3w + c);
    #pragma unroll
    for (int j = 0; j < 4; j++) accv += lrelu(z[j] * sc[j] + sh[j]) * aw[j];
  }
  float score = lrelu(accv + a3b[0]);
  int s = clampi(srcv[e], NN);
  int d = clampi(dstv[e], NN);
  float ee = __expf(-score) + (s == d ? 1.f : 0.f);
  eweight[e] = ee;
  atomicAdd(&rowsum[s], ee);
}

// ---------------------- aggregation: hprime[src] += edge_e * h[dst] (wave/edge)
__global__ void aggregate_kernel(const float* __restrict__ h, const float* __restrict__ eweight,
                                 const int* __restrict__ srcv, const int* __restrict__ dstv,
                                 float* __restrict__ hprime) {
  const int lane = threadIdx.x & 31;
  const int e = blockIdx.x * (blockDim.x >> 5) + (threadIdx.x >> 5);
  if (e >= EE) return;
  const int s = clampi(srcv[e], NN);
  const int d = clampi(dstv[e], NN);
  const float ee = eweight[e];
  v4f v = *(const v4f*)(h + (size_t)d * OUTC + lane * 4);
  float* out = hprime + (size_t)s * OUTC + lane * 4;
  atomicAdd(out + 0, ee * v[0]);
  atomicAdd(out + 1, ee * v[1]);
  atomicAdd(out + 2, ee * v[2]);
  atomicAdd(out + 3, ee * v[3]);
}

// ---------------------------------------- out = lrelu(hprime / rowsum')
__global__ void out_finalize_kernel(const float* __restrict__ hprime, const float* __restrict__ rowsum,
                                    float* __restrict__ out) {
  size_t i = (size_t)blockIdx.x * blockDim.x + threadIdx.x;
  if (i >= (size_t)NN * OUTC) return;
  float rs = rowsum[i >> 7];
  if (rs == 0.f) rs = 1.f;
  out[i] = lrelu(hprime[i] / rs);
}

extern "C" void kernel_launch(void* const* d_in, const int* in_sizes, int n_in,
                              void* d_out, int out_size, void* d_ws, size_t ws_size,
                              hipStream_t stream) {
  const float* x    = (const float*)d_in[0];
  const float* w    = (const float*)d_in[1];
  const float* a1w  = (const float*)d_in[2];
  const float* a1b  = (const float*)d_in[3];
  const float* bn1g = (const float*)d_in[4];
  const float* bn1b = (const float*)d_in[5];
  const float* a2w  = (const float*)d_in[6];
  const float* a2b  = (const float*)d_in[7];
  const float* bn2g = (const float*)d_in[8];
  const float* bn2b = (const float*)d_in[9];
  const float* a3w  = (const float*)d_in[10];
  const float* a3b  = (const float*)d_in[11];
  const int*   edge = (const int*)d_in[12];
  const int* srcv = edge;
  const int* dstv = edge + EE;

  // workspace layout (float units)
  float* wsf     = (float*)d_ws;
  float* hprime  = wsf;                    // 6,400,000
  float* rowsum  = wsf + 6400000;          // 50,000
  float* sum1    = wsf + 6450000;          // 64
  float* sq1     = sum1 + 64;              // 64
  float* sum2    = sum1 + 128;             // 32
  float* sq2     = sum1 + 160;             // 32   (zero region ends at 6,450,192)
  float* scale1  = wsf + 6450192;          // 64
  float* shift1  = scale1 + 64;            // 64
  float* scale2  = scale1 + 128;           // 32
  float* shift2  = scale1 + 160;           // 32
  float* h       = wsf + 6450384;          // 6,400,000
  float* z1      = wsf + 12850384;         // 51,200,000
  float* z2      = wsf + 64050384;         // 25,600,000
  float* eweight = wsf + 89650384;         // 800,000
  __bf16* xb     = (__bf16*)(wsf + 90450384); // 12,800,000 bf16
  __bf16* wT     = xb + (size_t)NN * INC;     // 32,768
  __bf16* a1T    = wT + INC * OUTC;           // 8,192
  __bf16* a2T    = a1T + OUTC * H1;           // 2,048

  const unsigned zeroN = 6450192u;
  zero_kernel<<<(zeroN + 255u) / 256u, 256, 0, stream>>>(wsf, zeroN);

  const size_t prepN = (size_t)NN * INC + INC * OUTC + OUTC * H1 + H1 * H2; // 12,843,008
  prep_kernel<<<(unsigned)((prepN + 255) / 256), 256, 0, stream>>>(x, w, a1w, a2w, xb, wT, a1T, a2T);

  node_gemm_kernel<<<(NN / 16) * (OUTC / 16) / 8, 256, 0, stream>>>(xb, wT, h);   // 3125 blocks

  edge_l1_kernel<<<(EE / 16) / 8, 256, 0, stream>>>(h, srcv, dstv, a1T, a1b, z1, sum1, sq1); // 6250

  bn_finalize_kernel<<<1, 64, 0, stream>>>(sum1, sq1, bn1g, bn1b, scale1, shift1, H1, 1.0f / EE);

  edge_l2_kernel<<<(EE / 16) / 8, 256, 0, stream>>>(z1, scale1, shift1, a2T, a2b, z2, sum2, sq2);

  bn_finalize_kernel<<<1, 64, 0, stream>>>(sum2, sq2, bn2g, bn2b, scale2, shift2, H2, 1.0f / EE);

  edge_score_kernel<<<EE / 256, 256, 0, stream>>>(z2, srcv, dstv, scale2, shift2, a3w, a3b, eweight, rowsum);

  aggregate_kernel<<<EE / 8, 256, 0, stream>>>(h, eweight, srcv, dstv, hprime);

  out_finalize_kernel<<<(NN * OUTC) / 256, 256, 0, stream>>>(hprime, rowsum, (float*)d_out);
}
